// Qwen2VLVisionBlock_26087631356169
// MI455X (gfx1250) — compile-verified, matrix operand order checked by reading, not run on hardware
//
#include <hip/hip_runtime.h>
#include <hip/hip_bf16.h>

#define S_   3072
#define D_   1280
#define H_   16
#define HD_  80
#define HDP_ 96     // HD padded to multiple of 32 for WMMA K-dim
#define FF_  5120

typedef _Float16 v16h __attribute__((ext_vector_type(16)));
typedef _Float16 v8h  __attribute__((ext_vector_type(8)));
typedef float    v8f  __attribute__((ext_vector_type(8)));

// gfx1250 async global->LDS copy path (ASYNCcnt-tracked), if this toolchain has it
#if __has_builtin(__builtin_amdgcn_global_load_async_to_lds_b128) && \
    __has_builtin(__builtin_amdgcn_s_wait_asynccnt)
#define ASYNC_LDS 1
typedef int v4i __attribute__((vector_size(16)));
typedef __attribute__((address_space(1))) v4i GV4;   // global int32x4
typedef __attribute__((address_space(3))) v4i LV4;   // LDS int32x4
#endif

// ---------------------------------------------------------------------------
// helpers
// ---------------------------------------------------------------------------
__device__ __forceinline__ v8f wmma_f16(v16h a, v16h b, v8f c) {
  // D = A(16x32 f16) * B(32x16 f16) + C(16x16 f32)
  return __builtin_amdgcn_wmma_f32_16x16x32_f16(false, a, false, b,
                                                (short)0, c, false, false);
}

// Build a 16-half fragment: halves 0..7 from p[0..7], halves 8..15 from p[16..23].
// Matches the 16-bit A/B fragment layout (lane holds K-pairs kb..kb+7, kb+16..kb+23).
__device__ __forceinline__ v16h mkfrag(const _Float16* p) {
  v8h a = *(const v8h*)p;
  v8h b = *(const v8h*)(p + 16);
  return __builtin_shufflevector(a, b, 0,1,2,3,4,5,6,7,8,9,10,11,12,13,14,15);
}

__device__ __forceinline__ v8f v8f_zero() {
  v8f z = {0.f,0.f,0.f,0.f,0.f,0.f,0.f,0.f};
  return z;
}

// ---------------------------------------------------------------------------
// seg[s] = searchsorted(cu_seqlens, s, side='right')
// ---------------------------------------------------------------------------
__global__ void seg_kernel(const int* __restrict__ cu, int* __restrict__ seg, int ncu) {
  int s = blockIdx.x * blockDim.x + threadIdx.x;
  if (s < S_) {
    int c = 0;
    for (int i = 0; i < ncu; ++i) c += (cu[i] <= s) ? 1 : 0;
    seg[s] = c;
  }
}

// ---------------------------------------------------------------------------
// LayerNorm over 1280 cols, one block (256 thr) per row, outputs f16
// ---------------------------------------------------------------------------
__global__ __launch_bounds__(256)
void ln_kernel(const float* __restrict__ x, const float* __restrict__ scale,
               const float* __restrict__ bias, _Float16* __restrict__ out) {
  __shared__ float red[8];
  const int row = blockIdx.x;
  const int tid = threadIdx.x, lane = tid & 31, w = tid >> 5;
  const float* xr = x + (long)row * D_;

  float vals[5];
  float s = 0.f;
  #pragma unroll
  for (int i = 0; i < 5; ++i) { vals[i] = xr[i * 256 + tid]; s += vals[i]; }
  #pragma unroll
  for (int off = 16; off; off >>= 1) s += __shfl_xor(s, off, 32);
  if (lane == 0) red[w] = s;
  __syncthreads();
  float tot = 0.f;
  #pragma unroll
  for (int j = 0; j < 8; ++j) tot += red[j];
  const float mu = tot * (1.0f / D_);
  __syncthreads();

  float vs = 0.f;
  #pragma unroll
  for (int i = 0; i < 5; ++i) { float d = vals[i] - mu; vs += d * d; }
  #pragma unroll
  for (int off = 16; off; off >>= 1) vs += __shfl_xor(vs, off, 32);
  if (lane == 0) red[w] = vs;
  __syncthreads();
  float tv = 0.f;
  #pragma unroll
  for (int j = 0; j < 8; ++j) tv += red[j];
  const float rstd = rsqrtf(tv * (1.0f / D_) + 1e-6f);

  _Float16* orow = out + (long)row * D_;
  #pragma unroll
  for (int i = 0; i < 5; ++i) {
    int col = i * 256 + tid;
    orow[col] = (_Float16)((vals[i] - mu) * rstd * scale[col] + bias[col]);
  }
}

// ---------------------------------------------------------------------------
// W[K][N] f32 -> Wt[N][K] f16
// ---------------------------------------------------------------------------
__global__ void castT_kernel(const float* __restrict__ W, _Float16* __restrict__ Wt,
                             int K, int N) {
  long idx = (long)blockIdx.x * blockDim.x + threadIdx.x;
  if (idx < (long)K * N) {
    int k = (int)(idx / N), n = (int)(idx % N);
    Wt[(long)n * K + k] = (_Float16)W[idx];
  }
}

// ---------------------------------------------------------------------------
// WMMA GEMM: C[M][N] = A[M][K](f16,rowmaj) * Bt[N][K](f16) + bias
// block tile 128x128, 8 waves of 32x64 (8 WMMA per K-step), K-step 32,
// double-buffered LDS staged via async global->LDS copies when available.
// ---------------------------------------------------------------------------
template<bool GELU, bool RESID, bool OUT32, bool OUT16>
__global__ __launch_bounds__(256)
void gemm_kernel(const _Float16* __restrict__ A, const _Float16* __restrict__ Bt,
                 const float* __restrict__ bias, const float* __restrict__ resid,
                 float* __restrict__ Cf, _Float16* __restrict__ Ch,
                 int M, int N, int K) {
  __shared__ __align__(16) _Float16 As[2][128 * 48];
  __shared__ __align__(16) _Float16 Bs[2][128 * 48];

  const int tid = threadIdx.x, lane = tid & 31, w = tid >> 5;
  const int wm = w >> 1, wn = w & 1;              // 4 x 2 wave grid, 32 x 64 tiles
  const int m0 = blockIdx.y * 128, n0 = blockIdx.x * 128;
  const int r = lane & 15, kb = (lane >> 4) << 3, hl = lane >> 4;

  v8f acc[2][4];
  #pragma unroll
  for (int mt = 0; mt < 2; ++mt)
    #pragma unroll
    for (int nt = 0; nt < 4; ++nt) acc[mt][nt] = v8f_zero();

  // stage one 128x32 A tile + 128x32 B tile into LDS buffer `buf`
  auto stage = [&](int buf, int k0) {
    #pragma unroll
    for (int c = 0; c < 2; ++c) {
      int idx = tid + c * 256;                    // 512 chunks of 8 halves
      int rr = idx >> 2, cc = (idx & 3) << 3;
      const _Float16* ga = &A[(long)(m0 + rr) * K + k0 + cc];
      const _Float16* gb = &Bt[(long)(n0 + rr) * K + k0 + cc];
      _Float16* la = &As[buf][rr * 48 + cc];
      _Float16* lb = &Bs[buf][rr * 48 + cc];
#ifdef ASYNC_LDS
      __builtin_amdgcn_global_load_async_to_lds_b128((GV4*)ga, (LV4*)la, 0, 0);
      __builtin_amdgcn_global_load_async_to_lds_b128((GV4*)gb, (LV4*)lb, 0, 0);
#else
      *(v8h*)la = *(const v8h*)ga;
      *(v8h*)lb = *(const v8h*)gb;
#endif
    }
  };

  const int nk = K >> 5;
  stage(0, 0);
  for (int it = 0; it < nk; ++it) {
#ifdef ASYNC_LDS
    __builtin_amdgcn_s_wait_asynccnt(0);
#endif
    __syncthreads();   // this wave's tile landed + everyone done reading other buf
    if (it + 1 < nk) stage((it + 1) & 1, (it + 1) << 5);

    const _Float16* as = As[it & 1];
    const _Float16* bs = Bs[it & 1];
    v16h af[2], bf[4];
    #pragma unroll
    for (int mt = 0; mt < 2; ++mt)
      af[mt] = mkfrag(&as[(wm * 32 + mt * 16 + r) * 48 + kb]);
    #pragma unroll
    for (int nt = 0; nt < 4; ++nt)
      bf[nt] = mkfrag(&bs[(wn * 64 + nt * 16 + r) * 48 + kb]);
    #pragma unroll
    for (int mt = 0; mt < 2; ++mt)
      #pragma unroll
      for (int nt = 0; nt < 4; ++nt)
        acc[mt][nt] = wmma_f16(af[mt], bf[nt], acc[mt][nt]);
  }

  // epilogue: C layout -> row m = vgpr_i + 8*hl, col n = lane&15
  #pragma unroll
  for (int mt = 0; mt < 2; ++mt)
    #pragma unroll
    for (int nt = 0; nt < 4; ++nt)
      #pragma unroll
      for (int i = 0; i < 8; ++i) {
        int gm = m0 + wm * 32 + mt * 16 + hl * 8 + i;
        int gn = n0 + wn * 64 + nt * 16 + r;
        float v = acc[mt][nt][i] + bias[gn];
        if (GELU)  v = v * (1.0f / (1.0f + __expf(-1.702f * v)));
        if (RESID) v += resid[(long)gm * N + gn];
        if (OUT32) Cf[(long)gm * N + gn] = v;
        if (OUT16) Ch[(long)gm * N + gn] = (_Float16)v;
      }
}

// ---------------------------------------------------------------------------
// RoPE + pack: qkv f32 [S][3*D] -> q/k f16 [H][S][96] (zero-padded),
//              vT f16 [H][80][S]
// ---------------------------------------------------------------------------
__global__ void rope_pack_kernel(const float* __restrict__ qkv,
                                 const float* __restrict__ rot,
                                 _Float16* __restrict__ q, _Float16* __restrict__ k,
                                 _Float16* __restrict__ vT) {
  int idx = blockIdx.x * blockDim.x + threadIdx.x;
  if (idx >= S_ * H_ * HDP_) return;
  int dp = idx % HDP_;
  int t  = idx / HDP_;
  int h  = t % H_;
  int s  = t / H_;

  _Float16 qo = (_Float16)0.f, ko = (_Float16)0.f;
  if (dp < HD_) {
    float fr = rot[s * (HD_ / 2) + (dp >> 1)];
    float c = cosf(fr), sn = sinf(fr);
    int base = s * (3 * D_) + h * HD_ + dp;
    float qv = qkv[base];
    float kv = qkv[base + D_];
    float vv = qkv[base + 2 * D_];
    int oth  = (dp < HD_ / 2) ? base + HD_ / 2 : base - HD_ / 2;
    float sgn = (dp < HD_ / 2) ? -1.f : 1.f;
    float qr = sgn * qkv[oth];
    float kr = sgn * qkv[oth + D_];
    qo = (_Float16)(qv * c + qr * sn);
    ko = (_Float16)(kv * c + kr * sn);
    vT[((long)(h * HD_ + dp)) * S_ + s] = (_Float16)vv;
  }
  long qoff = ((long)h * S_ + s) * HDP_ + dp;
  q[qoff] = qo;
  k[qoff] = ko;
}

// ---------------------------------------------------------------------------
// Flash attention: 1 wave per (head, 16-query tile). Online softmax kept in
// WMMA C-layout; segment mask from seg[]; disjoint key blocks skipped.
// ---------------------------------------------------------------------------
__global__ __launch_bounds__(32)
void attn_kernel(const _Float16* __restrict__ Q, const _Float16* __restrict__ Kt,
                 const _Float16* __restrict__ VT, const int* __restrict__ seg,
                 _Float16* __restrict__ attn_out) {
  __shared__ __align__(16) _Float16 pt[16 * 48];

  const int h = blockIdx.y, m0 = blockIdx.x * 16;
  const int lane = threadIdx.x;
  const int r = lane & 15, kb = (lane >> 4) << 3, hl = lane >> 4;
  const float SCALE = 0.11180339887498949f; // 1/sqrt(80)

  // Q fragments (K = 96 -> 3 steps of 32)
  v16h qf[3];
  const _Float16* qb = Q + ((long)h * S_ + m0 + r) * HDP_ + kb;
  #pragma unroll
  for (int ks = 0; ks < 3; ++ks) qf[ks] = mkfrag(qb + ks * 32);

  int sq[8];
  #pragma unroll
  for (int i = 0; i < 8; ++i) sq[i] = seg[m0 + hl * 8 + i];
  const int qs_lo = seg[m0], qs_hi = seg[m0 + 15];

  float rm[8], rl[8];
  #pragma unroll
  for (int i = 0; i < 8; ++i) { rm[i] = -3.0e38f; rl[i] = 0.f; }
  v8f o[5];
  #pragma unroll
  for (int d = 0; d < 5; ++d) o[d] = v8f_zero();

  for (int kb0 = 0; kb0 < S_; kb0 += 32) {
    int ks_lo = seg[kb0], ks_hi = seg[kb0 + 31];
    if (ks_hi < qs_lo || ks_lo > qs_hi) continue; // fully masked block

    // scores: 16x32 via two 16x16 tiles, K-dim = 96
    v8f sc[2]; sc[0] = v8f_zero(); sc[1] = v8f_zero();
    #pragma unroll
    for (int nt = 0; nt < 2; ++nt) {
      const _Float16* kbp = Kt + ((long)h * S_ + kb0 + nt * 16 + r) * HDP_ + kb;
      #pragma unroll
      for (int ks = 0; ks < 3; ++ks)
        sc[nt] = wmma_f16(qf[ks], mkfrag(kbp + ks * 32), sc[nt]);
    }

    const int sk0 = seg[kb0 + r];
    const int sk1 = seg[kb0 + 16 + r];

    float p0[8], p1[8];
    #pragma unroll
    for (int i = 0; i < 8; ++i) {
      float s0 = sc[0][i] * SCALE; if (sk0 != sq[i]) s0 = -1e30f;
      float s1 = sc[1][i] * SCALE; if (sk1 != sq[i]) s1 = -1e30f;
      float vmax = fmaxf(s0, s1);
      #pragma unroll
      for (int off = 1; off <= 8; off <<= 1)
        vmax = fmaxf(vmax, __shfl_xor(vmax, off, 32));
      float mnew = fmaxf(rm[i], vmax);
      float corr = __expf(rm[i] - mnew);
      float e0 = __expf(s0 - mnew);
      float e1 = __expf(s1 - mnew);
      float psum = e0 + e1;
      #pragma unroll
      for (int off = 1; off <= 8; off <<= 1)
        psum += __shfl_xor(psum, off, 32);
      rl[i] = rl[i] * corr + psum;
      rm[i] = mnew;
      #pragma unroll
      for (int d = 0; d < 5; ++d) o[d][i] *= corr;
      p0[i] = e0; p1[i] = e1;
    }

    // relayout P (C-layout) -> A-fragment via LDS (same-wave DS ops are in order)
    #pragma unroll
    for (int i = 0; i < 8; ++i) {
      pt[(hl * 8 + i) * 48 + r]      = (_Float16)p0[i];
      pt[(hl * 8 + i) * 48 + 16 + r] = (_Float16)p1[i];
    }
    v16h pf = mkfrag(&pt[r * 48 + kb]);

    // P(16x32) * V(32x80): 5 N-tiles, vT[h][d][s] rows contiguous along keys
    #pragma unroll
    for (int d = 0; d < 5; ++d) {
      const _Float16* vbp = VT + ((long)(h * HD_ + d * 16 + r)) * S_ + kb0 + kb;
      o[d] = wmma_f16(pf, mkfrag(vbp), o[d]);
    }
  }

  // normalize + scatter into [S][D] f16 (A operand of proj GEMM)
  #pragma unroll
  for (int d = 0; d < 5; ++d)
    #pragma unroll
    for (int i = 0; i < 8; ++i) {
      int m = hl * 8 + i;
      float val = o[d][i] / rl[i];
      attn_out[(long)(m0 + m) * D_ + h * HD_ + d * 16 + r] = (_Float16)val;
    }
}

// ---------------------------------------------------------------------------
// host launch
// ---------------------------------------------------------------------------
extern "C" void kernel_launch(void* const* d_in, const int* in_sizes, int n_in,
                              void* d_out, int out_size, void* d_ws, size_t ws_size,
                              hipStream_t stream) {
  const float* hidden = (const float*)d_in[0];
  const float* rope   = (const float*)d_in[1];
  const float* ln1s   = (const float*)d_in[2];
  const float* ln1b   = (const float*)d_in[3];
  const float* ln2s   = (const float*)d_in[4];
  const float* ln2b   = (const float*)d_in[5];
  const float* qkv_w  = (const float*)d_in[6];
  const float* qkv_b  = (const float*)d_in[7];
  const float* proj_w = (const float*)d_in[8];
  const float* proj_b = (const float*)d_in[9];
  const float* fc1_w  = (const float*)d_in[10];
  const float* fc1_b  = (const float*)d_in[11];
  const float* fc2_w  = (const float*)d_in[12];
  const float* fc2_b  = (const float*)d_in[13];
  const int*   cu     = (const int*)d_in[14];
  const int    ncu    = in_sizes[14];
  float* out = (float*)d_out;

  size_t off = 0;
  auto wsa = [&](size_t bytes) -> void* {
    void* p = (char*)d_ws + off;
    off += (bytes + 255) & ~(size_t)255;
    return p;
  };
  _Float16* h16    = (_Float16*)wsa((size_t)S_ * D_ * 2);
  _Float16* qkvT   = (_Float16*)wsa((size_t)(3 * D_) * D_ * 2);
  _Float16* projT  = (_Float16*)wsa((size_t)D_ * D_ * 2);
  _Float16* fc1T   = (_Float16*)wsa((size_t)FF_ * D_ * 2);
  _Float16* fc2T   = (_Float16*)wsa((size_t)D_ * FF_ * 2);
  float*    qkvbuf = (float*)wsa((size_t)S_ * 3 * D_ * 4);
  _Float16* q16    = (_Float16*)wsa((size_t)H_ * S_ * HDP_ * 2);
  _Float16* k16    = (_Float16*)wsa((size_t)H_ * S_ * HDP_ * 2);
  _Float16* vT16   = (_Float16*)wsa((size_t)H_ * HD_ * S_ * 2);
  int*      seg    = (int*)wsa((size_t)S_ * 4);
  _Float16* attn16 = (_Float16*)wsa((size_t)S_ * D_ * 2);
  float*    x1     = (float*)wsa((size_t)S_ * D_ * 4);
  _Float16* h2_16  = (_Float16*)wsa((size_t)S_ * D_ * 2);
  _Float16* g16    = (_Float16*)wsa((size_t)S_ * FF_ * 2);

  // segment ids + LN1
  seg_kernel<<<(S_ + 255) / 256, 256, 0, stream>>>(cu, seg, ncu);
  ln_kernel<<<S_, 256, 0, stream>>>(hidden, ln1s, ln1b, h16);

  // weight cast + transpose to f16 [N][K]
  castT_kernel<<<(D_ * 3 * D_ + 255) / 256, 256, 0, stream>>>(qkv_w, qkvT, D_, 3 * D_);
  castT_kernel<<<(D_ * D_ + 255) / 256, 256, 0, stream>>>(proj_w, projT, D_, D_);
  castT_kernel<<<(D_ * FF_ + 255) / 256, 256, 0, stream>>>(fc1_w, fc1T, D_, FF_);
  castT_kernel<<<(FF_ * D_ + 255) / 256, 256, 0, stream>>>(fc2_w, fc2T, FF_, D_);

  // QKV projection
  gemm_kernel<false, false, true, false>
      <<<dim3(3 * D_ / 128, S_ / 128), 256, 0, stream>>>(
          h16, qkvT, qkv_b, nullptr, qkvbuf, nullptr, S_, 3 * D_, D_);

  // RoPE + head-major packing
  rope_pack_kernel<<<(S_ * H_ * HDP_ + 255) / 256, 256, 0, stream>>>(
      qkvbuf, rope, q16, k16, vT16);

  // flash attention
  attn_kernel<<<dim3(S_ / 16, H_), 32, 0, stream>>>(q16, k16, vT16, seg, attn16);

  // output projection + residual -> x1
  gemm_kernel<false, true, true, false>
      <<<dim3(D_ / 128, S_ / 128), 256, 0, stream>>>(
          attn16, projT, proj_b, hidden, x1, nullptr, S_, D_, D_);

  // LN2
  ln_kernel<<<S_, 256, 0, stream>>>(x1, ln2s, ln2b, h2_16);

  // FC1 + quickGELU (f16 out only)
  gemm_kernel<true, false, false, true>
      <<<dim3(FF_ / 128, S_ / 128), 256, 0, stream>>>(
          h2_16, fc1T, fc1_b, nullptr, nullptr, g16, S_, FF_, D_);

  // FC2 + residual -> final output
  gemm_kernel<false, true, true, false>
      <<<dim3(D_ / 128, S_ / 128), 256, 0, stream>>>(
          g16, fc2T, fc2_b, x1, out, nullptr, S_, D_, FF_);
}
